// Agent_74826920231016
// MI455X (gfx1250) — matrix operand order repaired and checked by reading, not compiled, not run on hardware
//
#include <hip/hip_runtime.h>

// ---------------------------------------------------------------------------
// LSTM(1024) x 4096 steps, batch 1, decoder folded into recurrence.
//   W_eff = W_hh + W_ih @ W_dec   (bf16), b_eff = b_ih + b_hh + W_ih @ b_dec
//   gates_t = W_eff @ h_{t-1} + b_eff   (t>=1);  gates_0 = W_ih@x + b_ih + b_hh
// 32 persistent WGs x 256 thr; each WG owns 32 hidden units (128 gate rows).
// Weights staged global->LDS via TDM once, then pinned in VGPRs (256/lane).
// Steady-state step: 32 x v_wmma_f32_16x16x32_bf16 + 64 ds_load_b128 per wave.
// Decoder outputs computed in a post-pass from the stored h history.
// ---------------------------------------------------------------------------

#define H        1024
#define G4       4096
#define STEPS    4096
#define REC_WGS  32
#define UNITS    32      // hidden units per WG
#define ROWS     128     // 4 gates * UNITS
#define KCH      32      // 1024 / 32

typedef __attribute__((ext_vector_type(8)))  float        v8f;
typedef __attribute__((ext_vector_type(16))) __bf16       v16bf;
typedef __attribute__((ext_vector_type(8)))  __bf16       v8bf;
typedef __attribute__((ext_vector_type(4)))  unsigned int v4u;
typedef __attribute__((ext_vector_type(8)))  int          v8i;
typedef __attribute__((ext_vector_type(4)))  int          v4i;

union V16 { v16bf v; v8bf h[2]; };

__device__ __forceinline__ unsigned short f2bf(float f) {
    unsigned u = __float_as_uint(f);
    u += 0x7fffu + ((u >> 16) & 1u);       // round-to-nearest-even
    return (unsigned short)(u >> 16);
}
__device__ __forceinline__ float sigm(float x) {
    return 1.0f / (1.0f + __expf(-x));
}

// ---------------------------------------------------------------------------
// Prep 1: b_eff, g0 (step-0 gate preactivations), zero h state + barrier ctr.
// ---------------------------------------------------------------------------
__global__ void Agent_prep_vec(const float* __restrict__ W_ih,
                               const float* __restrict__ b_ih,
                               const float* __restrict__ b_hh,
                               const float* __restrict__ b_dec,
                               const float* __restrict__ x,
                               float* __restrict__ beff,
                               float* __restrict__ g0,
                               float* __restrict__ hbuf,
                               unsigned* __restrict__ ctr) {
    int r = blockIdx.x * blockDim.x + threadIdx.x;       // 0..4095
    if (r < G4) {
        float base = b_ih[r] + b_hh[r];
        float sb = 0.f, sx = 0.f;
        #pragma unroll
        for (int j = 0; j < 11; ++j) {
            float w = W_ih[r * 11 + j];
            sb += w * b_dec[j];
            sx += w * x[j];
        }
        beff[r] = base + sb;
        g0[r]   = base + sx;
        if (r < H) hbuf[r] = 0.f;
        if (r == 0) *ctr = 0u;
    }
}

// ---------------------------------------------------------------------------
// Prep 2: W_eff = W_hh + W_ih @ W_dec, stored bf16 row-major [4096,1024].
// ---------------------------------------------------------------------------
__global__ void Agent_prep_weff(const float* __restrict__ W_hh,
                                const float* __restrict__ W_ih,
                                const float* __restrict__ W_dec,
                                unsigned short* __restrict__ weff) {
    int e0 = (blockIdx.x * blockDim.x + threadIdx.x) * 4;
    #pragma unroll
    for (int k = 0; k < 4; ++k) {
        int e = e0 + k;                    // < 4096*1024
        int r = e >> 10, c = e & 1023;
        float acc = W_hh[e];
        #pragma unroll
        for (int j = 0; j < 11; ++j)
            acc += W_ih[r * 11 + j] * W_dec[j * H + c];
        weff[e] = f2bf(acc);
    }
}

// ---------------------------------------------------------------------------
// Optional TDM staging of a [rows x cols] bf16 tile into LDS.
// ---------------------------------------------------------------------------
#if __has_builtin(__builtin_amdgcn_tensor_load_to_lds) && \
    __has_builtin(__builtin_amdgcn_s_wait_tensorcnt)
#define USE_TDM 1
__device__ __forceinline__ void tdm_load_tile(const unsigned short* gsrc,
                                              unsigned lds_addr,
                                              unsigned rows, unsigned cols) {
    unsigned long long ga = (unsigned long long)(size_t)gsrc;
    v4u g0d; v8i g1d; v4i g2d = {0,0,0,0}; v4i g3d = {0,0,0,0};
    g0d[0] = 1u;                                       // count = 1 (valid)
    g0d[1] = lds_addr;                                 // lds_addr
    g0d[2] = (unsigned)(ga & 0xFFFFFFFFull);           // global_addr[31:0]
    g0d[3] = (unsigned)((ga >> 32) & 0x01FFFFFFull) | (2u << 30); // type=2
    g1d[0] = (int)0x00010000u;                         // data_size = 2 bytes
    g1d[1] = (int)((cols & 0xFFFFu) << 16);            // tensor_dim0 lo
    g1d[2] = (int)(((cols >> 16) & 0xFFFFu) | ((rows & 0xFFFFu) << 16));
    g1d[3] = (int)(((rows >> 16) & 0xFFFFu) | ((cols & 0xFFFFu) << 16)); // tile_dim0
    g1d[4] = (int)(rows & 0xFFFFu);                    // tile_dim1
    g1d[5] = (int)cols;                                // tensor_dim0_stride
    g1d[6] = 0; g1d[7] = 0;
#if defined(__clang_major__) && (__clang_major__ >= 23)
    v8i g4d = {0,0,0,0,0,0,0,0};
    __builtin_amdgcn_tensor_load_to_lds(g0d, g1d, g2d, g3d, g4d, 0);
#else
    __builtin_amdgcn_tensor_load_to_lds(g0d, g1d, g2d, g3d, 0);
#endif
}
#else
#define USE_TDM 0
#endif

// ---------------------------------------------------------------------------
// Persistent recurrence kernel.
// ---------------------------------------------------------------------------
__global__ __launch_bounds__(256, 1)
void Agent_lstm_persistent(const unsigned short* __restrict__ weff,
                           const float* __restrict__ beff,
                           const float* __restrict__ g0,
                           float* __restrict__ hbuf,
                           float* __restrict__ hhist,
                           unsigned* __restrict__ ctr) {
    __shared__ __align__(16) unsigned short wlds[ROWS * H];  // 256 KB bf16
    __shared__ __align__(16) unsigned short shb[H];          // h in bf16
    __shared__ float sgates[ROWS];
    __shared__ float sbeff[ROWS];

    const int tid  = threadIdx.x;
    const int wave = tid >> 5;
    const int lane = tid & 31;
    const int ub   = blockIdx.x * UNITS;                 // first hidden unit

    // ---- stage the weight slice (rows {gate*1024 + ub .. +32}) into LDS ----
#if USE_TDM
    if (tid < 32) {                                      // wave 0 issues DMA
        #pragma unroll
        for (int gate = 0; gate < 4; ++gate) {
            const unsigned short* src = weff + (size_t)(gate * H + ub) * H;
            unsigned lds_addr = (unsigned)(size_t)(void*)&wlds[gate * UNITS * H];
            tdm_load_tile(src, lds_addr, UNITS, H);
        }
        __builtin_amdgcn_s_wait_tensorcnt(0);
    }
#else
    for (int lr = tid >> 1; lr < ROWS; lr += 128) {      // 2 threads per row
        int gate = lr >> 5, u = lr & 31;
        const uint4* src = (const uint4*)(weff + (size_t)(gate * H + ub + u) * H);
        uint4* dst = (uint4*)&wlds[lr * H];
        int half = tid & 1;
        for (int i = half * 64; i < half * 64 + 64; ++i) dst[i] = src[i];
    }
#endif
    for (int lr = tid; lr < ROWS; lr += 256) {
        int gate = lr >> 5, u = lr & 31;
        sbeff[lr] = beff[gate * H + ub + u];
    }
    __syncthreads();

    const v8bf* w8 = (const v8bf*)wlds;
    const v8bf* h8 = (const v8bf*)shb;
    const int lr16 = wave * 16 + (lane & 15);            // this lane's A row
    const int selA = (lane >> 4) ? 16 : 0;               // K sub-block for A
    const int selB = (lane >> 4) ? 16 : 0;               // K sub-block for B

    // ---- pin this wave's A strip (16 rows x 1024 K, bf16) in registers ----
    // 32 tiles x 32B/lane = 1KB/lane = 256 VGPRs, resident across all steps.
    V16 aReg[KCH];
    #pragma unroll
    for (int kc = 0; kc < KCH; ++kc) {
        const int k0 = kc * 32;
        aReg[kc].h[0] = w8[(lr16 * H + k0 + selA) >> 3];
        aReg[kc].h[1] = w8[(lr16 * H + k0 + selA + 8) >> 3];
    }

    float creg = 0.f;                                    // cell state (tid<32)

    for (int t = 0; t < STEPS; ++t) {
        if (t == 0) {
            for (int lr = tid; lr < ROWS; lr += 256) {
                int gate = lr >> 5, u = lr & 31;
                sgates[lr] = g0[gate * H + ub + u];
            }
            __syncthreads();
        } else {
            // h (f32, global) -> bf16 LDS, vectorized: 4 elems/thread
            {
                const int k4 = tid * 4;
                const float4 hv = *(const float4*)&hbuf[k4];
                uint2 pk;
                pk.x = (unsigned)f2bf(hv.x) | ((unsigned)f2bf(hv.y) << 16);
                pk.y = (unsigned)f2bf(hv.z) | ((unsigned)f2bf(hv.w) << 16);
                *(uint2*)&shb[k4] = pk;
            }
            __syncthreads();

            // Two accumulator chains to break WMMA->WMMA RAW dependency.
            v8f acc0 = {0.f,0.f,0.f,0.f,0.f,0.f,0.f,0.f};
            v8f acc1 = {0.f,0.f,0.f,0.f,0.f,0.f,0.f,0.f};
            #pragma unroll
            for (int kc = 0; kc < KCH; kc += 2) {
                V16 b0, b1;
                b0.h[0] = h8[(kc * 32 + selB) >> 3];
                b0.h[1] = h8[(kc * 32 + selB + 8) >> 3];
                b1.h[0] = h8[(kc * 32 + 32 + selB) >> 3];
                b1.h[1] = h8[(kc * 32 + 32 + selB + 8) >> 3];
                acc0 = __builtin_amdgcn_wmma_f32_16x16x32_bf16(
                           false, aReg[kc].v,     false, b0.v, (short)0, acc0, false, false);
                acc1 = __builtin_amdgcn_wmma_f32_16x16x32_bf16(
                           false, aReg[kc + 1].v, false, b1.v, (short)0, acc1, false, false);
            }
            v8f acc = acc0 + acc1;

            // C layout: VGPR v -> row (v + 8*(lane>=16)), col = lane%16.
            if (lane == 0 || lane == 16) {
                int base = wave * 16 + ((lane >> 4) << 3);
                #pragma unroll
                for (int v = 0; v < 8; ++v)
                    sgates[base + v] = acc[v] + sbeff[base + v];
            }
            __syncthreads();
        }

        // ---- pointwise LSTM update: one thread per owned hidden unit ----
        if (tid < UNITS) {
            float ig = sigm(sgates[tid]);
            float fg = sigm(sgates[UNITS + tid]);
            float gg = tanhf(sgates[2 * UNITS + tid]);
            float og = sigm(sgates[3 * UNITS + tid]);
            creg = fg * creg + ig * gg;
            float hv = og * tanhf(creg);
            hbuf[ub + tid] = hv;
            hhist[(size_t)t * H + ub + tid] = hv;
        }

        // ---- grid barrier: release arrival, acquire polling + s_sleep ----
        __syncthreads();
        if (tid == 0) {
            __threadfence();
            __hip_atomic_fetch_add(ctr, 1u, __ATOMIC_RELEASE,
                                   __HIP_MEMORY_SCOPE_AGENT);
            const unsigned target = (unsigned)(t + 1) * (unsigned)gridDim.x;
            while (__hip_atomic_load(ctr, __ATOMIC_ACQUIRE,
                                     __HIP_MEMORY_SCOPE_AGENT) < target)
                __builtin_amdgcn_s_sleep(2);
        }
        __syncthreads();
    }
}

// ---------------------------------------------------------------------------
// Post-pass decoder: out[t,:] = hhist[t,:] @ W_dec.T + b_dec.
// ---------------------------------------------------------------------------
__global__ void Agent_decoder(const float* __restrict__ hhist,
                              const float* __restrict__ W_dec,
                              const float* __restrict__ b_dec,
                              float* __restrict__ out) {
    __shared__ float red[256];
    const int t = blockIdx.x, tid = threadIdx.x;
    float acc[11];
    #pragma unroll
    for (int n = 0; n < 11; ++n) acc[n] = 0.f;
    for (int k = tid; k < H; k += 256) {
        float hv = hhist[(size_t)t * H + k];
        #pragma unroll
        for (int n = 0; n < 11; ++n) acc[n] += hv * W_dec[n * H + k];
    }
    for (int n = 0; n < 11; ++n) {
        red[tid] = acc[n];
        __syncthreads();
        for (int s = 128; s > 0; s >>= 1) {
            if (tid < s) red[tid] += red[tid + s];
            __syncthreads();
        }
        if (tid == 0) out[t * 11 + n] = red[0] + b_dec[n];
        __syncthreads();
    }
}

// ---------------------------------------------------------------------------
extern "C" void kernel_launch(void* const* d_in, const int* in_sizes, int n_in,
                              void* d_out, int out_size, void* d_ws, size_t ws_size,
                              hipStream_t stream) {
    (void)in_sizes; (void)n_in; (void)out_size; (void)ws_size;
    const float* x     = (const float*)d_in[0];
    const float* W_ih  = (const float*)d_in[1];
    const float* W_hh  = (const float*)d_in[2];
    const float* b_ih  = (const float*)d_in[3];
    const float* b_hh  = (const float*)d_in[4];
    const float* W_dec = (const float*)d_in[5];
    const float* b_dec = (const float*)d_in[6];
    float* out = (float*)d_out;

    char* ws = (char*)d_ws;
    unsigned short* weff = (unsigned short*)ws;            //  8 MB  bf16 W_eff
    float* beff  = (float*)(ws + 8388608);                 // 16 KB
    float* g0    = (float*)(ws + 8404992);                 // 16 KB
    float* hbuf  = (float*)(ws + 8421376);                 //  4 KB
    unsigned* ctr = (unsigned*)(ws + 8425472);             //  barrier counter
    float* hhist = (float*)(ws + 8429568);                 // 16 MB h history

    Agent_prep_vec<<<16, 256, 0, stream>>>(W_ih, b_ih, b_hh, b_dec, x,
                                           beff, g0, hbuf, ctr);
    Agent_prep_weff<<<4096, 256, 0, stream>>>(W_hh, W_ih, W_dec, weff);
    Agent_lstm_persistent<<<REC_WGS, 256, 0, stream>>>(weff, beff, g0,
                                                       hbuf, hhist, ctr);
    Agent_decoder<<<STEPS, 256, 0, stream>>>(hhist, W_dec, b_dec, out);
}
